// SwinTransformerBlock_3865470566403
// MI455X (gfx1250) — matrix operand order repaired and checked by reading, not compiled, not run on hardware
//
#include <hip/hip_runtime.h>
#include <hip/hip_bf16.h>

typedef __attribute__((ext_vector_type(16))) _Float16 v16h;
typedef __attribute__((ext_vector_type(8)))  float    v8f;

#define WMMA_F16(A,B,C) __builtin_amdgcn_wmma_f32_16x16x32_f16(false,(A),false,(B),(short)0,(C),false,false)

// ---- constants ----
#define BB   16
#define HH   112
#define CC   96
#define NHD  3
#define WS   7
#define SS   3
#define NTK  49           // tokens per window
#define NWIN 4096         // B * (112/7)^2
#define MTOT 200704       // B*H*W  (divisible by 16)
#define HID  384
#define QKVN 288

// A-matrix 16x32 f16 fragment: element e of v16h for lane (l16,half) holds K:
__device__ __forceinline__ int a_kofs(int e, int hh) {
    int vg = e >> 1;
    return ((vg < 4) ? (2 * vg) : (8 + 2 * vg)) + 8 * hh + (e & 1);
}
// B-matrix 32x16 f16 fragment: lane = N column; element e holds K:
__device__ __forceinline__ int b_kofs(int e, int hh) {
    return ((e < 8) ? e : (8 + e)) + 8 * hh;
}

// ---------------- TDM weight staging ----------------
#ifndef __has_builtin
#define __has_builtin(x) 0
#endif
#if __has_builtin(__builtin_amdgcn_tensor_load_to_lds) && __has_builtin(__builtin_amdgcn_s_wait_tensorcnt)
#define HAVE_TDM 1
typedef __attribute__((ext_vector_type(4))) unsigned int u32x4;
typedef __attribute__((ext_vector_type(8))) int i32x8;
typedef __attribute__((ext_vector_type(4))) int i32x4;
#else
#define HAVE_TDM 0
#endif

// Stage nelems f16 weights (contiguous) from global into LDS.
__device__ __forceinline__ void stage_weights(_Float16* wlds, const _Float16* src, int nelems) {
#if HAVE_TDM
    if (threadIdx.x < 32) {   // one wave issues the DMA (EXEC-independent op)
        unsigned long long ga = (unsigned long long)(uintptr_t)src;
        unsigned ldsOff = (unsigned)(uintptr_t)wlds;   // low 32 bits of LDS aperture addr = LDS byte offset
        unsigned ne = (unsigned)nelems;
        // D# group0: count=1 | lds_addr | global_addr[56:0] | type=2
        u32x4 g0 = { 1u,
                     ldsOff,
                     (unsigned)ga,
                     (unsigned)((ga >> 32) & 0x1ffffffu) | (2u << 30) };
        // D# group1: data_size=1(2B); tensor_dim0=tile_dim0=nelems; dims1..=1; stride0=stride1=nelems
        i32x8 g1 = { (int)(1u << 16),                       // data_size=2B at bits[17:16]
                     (int)((ne & 0xffffu) << 16),           // tensor_dim0[15:0] @ bits 63:48
                     (int)((ne >> 16) | (1u << 16)),        // tensor_dim0[31:16], tensor_dim1=1
                     (int)((ne & 0xffffu) << 16),           // tile_dim0 @ bits 127:112
                     1,                                     // tile_dim1=1, tile_dim2=0
                     (int)ne,                               // tensor_dim0_stride[31:0]
                     (int)((ne & 0xffffu) << 16),           // stride0[47:32]=0, stride1[15:0]
                     (int)(ne >> 16) };                     // stride1[47:16]
        i32x4 gz4 = { 0, 0, 0, 0 };
        i32x8 gz8 = { 0, 0, 0, 0, 0, 0, 0, 0 };
        __builtin_amdgcn_tensor_load_to_lds(g0, g1, gz4, gz4, gz8, 0);
        __builtin_amdgcn_s_wait_tensorcnt(0);
    }
#else
    for (int i = threadIdx.x * 8; i < nelems; i += blockDim.x * 8)
        *reinterpret_cast<int4*>(wlds + i) = *reinterpret_cast<const int4*>(src + i);
#endif
}

// ---------- f32 -> f16 convert (weights) ----------
__global__ void cvt_f16_kernel(const float* __restrict__ in, _Float16* __restrict__ out, int n) {
    int i = blockIdx.x * blockDim.x + threadIdx.x;
    if (i < n) out[i] = (_Float16)in[i];
}

// ---------- LN1 + cyclic shift + window partition (f32 -> f16) ----------
__global__ void ln1_shift_window_kernel(const float* __restrict__ x,
                                        const float* __restrict__ g,
                                        const float* __restrict__ b,
                                        _Float16* __restrict__ xw) {
    int token = blockIdx.x * (blockDim.x >> 5) + (threadIdx.x >> 5);
    int lane  = threadIdx.x & 31;
    if (token >= MTOT) return;
    const float* row = x + (long)token * CC;
    float v0 = row[lane], v1 = row[lane + 32], v2 = row[lane + 64];
    float s = v0 + v1 + v2;
    #pragma unroll
    for (int m = 1; m < 32; m <<= 1) s += __shfl_xor(s, m, 32);
    float mean = s * (1.0f / 96.0f);
    float d0 = v0 - mean, d1 = v1 - mean, d2 = v2 - mean;
    float q = d0 * d0 + d1 * d1 + d2 * d2;
    #pragma unroll
    for (int m = 1; m < 32; m <<= 1) q += __shfl_xor(q, m, 32);
    float rstd = rsqrtf(q * (1.0f / 96.0f) + 1e-5f);
    int bi = token / (HH * HH), hw = token % (HH * HH);
    int h = hw / HH, w = hw % HH;
    int hs = (h + HH - SS) % HH, ws = (w + HH - SS) % HH;
    int wrow = ((bi * 256 + (hs / WS) * 16 + (ws / WS)) * NTK) + (hs % WS) * WS + (ws % WS);
    _Float16* o = xw + (long)wrow * CC;
    o[lane]      = (_Float16)(d0 * rstd * g[lane]      + b[lane]);
    o[lane + 32] = (_Float16)(d1 * rstd * g[lane + 32] + b[lane + 32]);
    o[lane + 64] = (_Float16)(d2 * rstd * g[lane + 64] + b[lane + 64]);
}

// ---------- LN2 (f32 -> f16, token order) ----------
__global__ void ln2_kernel(const float* __restrict__ x1,
                           const float* __restrict__ g,
                           const float* __restrict__ b,
                           _Float16* __restrict__ xn2) {
    int token = blockIdx.x * (blockDim.x >> 5) + (threadIdx.x >> 5);
    int lane  = threadIdx.x & 31;
    if (token >= MTOT) return;
    const float* row = x1 + (long)token * CC;
    float v0 = row[lane], v1 = row[lane + 32], v2 = row[lane + 64];
    float s = v0 + v1 + v2;
    #pragma unroll
    for (int m = 1; m < 32; m <<= 1) s += __shfl_xor(s, m, 32);
    float mean = s * (1.0f / 96.0f);
    float d0 = v0 - mean, d1 = v1 - mean, d2 = v2 - mean;
    float q = d0 * d0 + d1 * d1 + d2 * d2;
    #pragma unroll
    for (int m = 1; m < 32; m <<= 1) q += __shfl_xor(q, m, 32);
    float rstd = rsqrtf(q * (1.0f / 96.0f) + 1e-5f);
    _Float16* o = xn2 + (long)token * CC;
    o[lane]      = (_Float16)(d0 * rstd * g[lane]      + b[lane]);
    o[lane + 32] = (_Float16)(d1 * rstd * g[lane + 32] + b[lane + 32]);
    o[lane + 64] = (_Float16)(d2 * rstd * g[lane + 64] + b[lane + 64]);
}

// ---------- unified strip GEMM with LDS-staged weights ----------
// C[M,NTOT] = A[M,K] * Wt^T (+ epilogue);  Wt is [NTOT,K] row-major so B[k][n] = Wt[n*K+k].
// EPI: 0=qkv(store f16 + bias), 1=proj(window-reverse+unshift+residual -> f32),
//      2=fc1(bias+GELU -> f16), 3=fc2(bias+residual -> f32)
template<int KSTEPS, int NTOT, int NCHUNK, int EPI>
__global__ void gemm_kernel(const _Float16* __restrict__ A,
                            const _Float16* __restrict__ Wt,
                            const float* __restrict__ bias,
                            const float* __restrict__ aux,   // shortcut (proj) / x1 (fc2)
                            void* __restrict__ outv) {
    constexpr int K = KSTEPS * 32;
    extern __shared__ _Float16 wlds[];
    int wave = threadIdx.x >> 5, lane = threadIdx.x & 31;
    int hh = lane >> 4, l16 = lane & 15;
    int m0 = (blockIdx.x * 8 + wave) * 16;   // grid sized exactly: m0 < MTOT always

    // resident A fragments for this 16-row strip
    v16h a[KSTEPS];
    const _Float16* arow = A + (long)(m0 + l16) * K;
    #pragma unroll
    for (int s = 0; s < KSTEPS; ++s)
        #pragma unroll
        for (int e = 0; e < 16; ++e)
            a[s][e] = arow[32 * s + a_kofs(e, hh)];

    #pragma unroll 1
    for (int p = 0; p < NTOT / NCHUNK; ++p) {
        if (p) __syncthreads();                         // protect LDS before overwrite
        stage_weights(wlds, Wt + (long)p * NCHUNK * K, NCHUNK * K);
        __syncthreads();

        #pragma unroll 1
        for (int n0l = 0; n0l < NCHUNK; n0l += 16) {
            v8f acc = {};
            #pragma unroll
            for (int s = 0; s < KSTEPS; ++s) {
                v16h bf;
                const _Float16* wr = wlds + (n0l + l16) * K + 32 * s;
                #pragma unroll
                for (int e = 0; e < 16; ++e) bf[e] = wr[b_kofs(e, hh)];
                acc = WMMA_F16(a[s], bf, acc);
            }
            int nn = p * NCHUNK + n0l + l16;
            float bv = bias[nn];
            #pragma unroll
            for (int r = 0; r < 8; ++r) {
                int mm = m0 + r + 8 * hh;
                float v = acc[r] + bv;
                if constexpr (EPI == 0) {
                    ((_Float16*)outv)[(long)mm * NTOT + nn] = (_Float16)v;
                } else if constexpr (EPI == 1) {
                    int w = mm / NTK, t = mm % NTK;
                    int bi = w / 256, widx = w % 256;
                    int hs = (widx / 16) * WS + t / WS;
                    int ws = (widx % 16) * WS + t % WS;
                    int h = (hs + SS) % HH, wc = (ws + SS) % HH;   // reverse shift
                    long oi = ((long)bi * HH * HH + h * HH + wc) * CC + nn;
                    ((float*)outv)[oi] = aux[oi] + v;
                } else if constexpr (EPI == 2) {
                    v = 0.5f * v * (1.0f + erff(v * 0.70710678118654752f));  // exact GELU
                    ((_Float16*)outv)[(long)mm * NTOT + nn] = (_Float16)v;
                } else {
                    long oi = (long)mm * NTOT + nn;
                    ((float*)outv)[oi] = aux[oi] + v;
                }
            }
        }
    }
}

// ---------- windowed attention: one block per window, 12 waves = 3 heads x 4 strips ----------
__global__ void __launch_bounds__(384) attn_kernel(const _Float16* __restrict__ qkv,
                                                   const float* __restrict__ rel_bias,   // [169,3]
                                                   _Float16* __restrict__ attn_out) {    // [M,96] f16
    int w = blockIdx.x;
    int tid = threadIdx.x;
    int wave = tid >> 5, lane = tid & 31;
    int head = wave >> 2, wv = wave & 3;
    int hh = lane >> 4, l16 = lane & 15;

    __shared__ _Float16 qklds[64 * 192];      // padded Q|K rows (cols 0..95 = Q, 96..191 = K)
    __shared__ _Float16 vt[96 * 64];          // V transposed: vt[c][t] = V[t][c]
    __shared__ _Float16 plds[12][16 * 64];    // P staging per wave

    const _Float16* base = qkv + (long)w * NTK * QKVN;

    // ---- stage Q,K zero-padded to 64 rows (vectorized, coalesced) ----
    for (int idx = tid; idx < 64 * 24; idx += 384) {           // 24 x int4 chunks per row
        int t = idx / 24, c8 = (idx % 24) * 8;
        int4 val = make_int4(0, 0, 0, 0);
        if (t < NTK) val = *reinterpret_cast<const int4*>(base + (long)t * QKVN + c8);
        *reinterpret_cast<int4*>(&qklds[t * 192 + c8]) = val;
    }
    // ---- stage V transposed (rows >= 49 zero) ----
    for (int idx = tid; idx < 64 * 12; idx += 384) {           // 12 x int4 chunks per row
        int t = idx / 12, c8 = (idx % 12) * 8;
        int4 val = make_int4(0, 0, 0, 0);
        if (t < NTK) val = *reinterpret_cast<const int4*>(base + (long)t * QKVN + 192 + c8);
        _Float16 tmp[8];
        *reinterpret_cast<int4*>(tmp) = val;
        #pragma unroll
        for (int j2 = 0; j2 < 8; ++j2) vt[(c8 + j2) * 64 + t] = tmp[j2];
    }
    __syncthreads();

    int m0 = wv * 16;

    // Q fragment (16 rows x K=32) — unguarded ds loads
    v16h aq;
    #pragma unroll
    for (int e = 0; e < 16; ++e)
        aq[e] = qklds[(m0 + l16) * 192 + head * 32 + a_kofs(e, hh)];

    // S = Q K^T  (4 column tiles of 16)
    float sv[4][8];
    #pragma unroll
    for (int j = 0; j < 4; ++j) {
        v16h bk;
        #pragma unroll
        for (int e = 0; e < 16; ++e)
            bk[e] = qklds[(j * 16 + l16) * 192 + 96 + head * 32 + b_kofs(e, hh)];
        v8f acc = {};
        acc = WMMA_F16(aq, bk, acc);
        #pragma unroll
        for (int r = 0; r < 8; ++r) sv[j][r] = acc[r];
    }

    // scale + relative position bias + shift mask (static formulas)
    int widx = w % 256;
    int wi = widx / 16, wj = widx % 16;
    const float scale = 0.17677669529663687f;  // 32^-0.5
    #pragma unroll
    for (int j = 0; j < 4; ++j) {
        int jt = j * 16 + l16;
        #pragma unroll
        for (int r = 0; r < 8; ++r) {
            int it = m0 + r + 8 * hh;
            float s = sv[j][r] * scale;
            if (it < NTK && jt < NTK) {
                int ri = it / WS, ci = it % WS, rj = jt / WS, cj = jt % WS;
                int rel = (ri - rj + 6) * 13 + (ci - cj + 6);
                s += rel_bias[rel * NHD + head];
                int hi = wi * WS + ri, vi = wj * WS + ci;
                int hj = wi * WS + rj, vj = wj * WS + cj;
                int regi = 3 * (hi < 105 ? 0 : (hi < 109 ? 1 : 2)) + (vi < 105 ? 0 : (vi < 109 ? 1 : 2));
                int regj = 3 * (hj < 105 ? 0 : (hj < 109 ? 1 : 2)) + (vj < 105 ? 0 : (vj < 109 ? 1 : 2));
                if (regi != regj) s -= 100.0f;
            } else {
                s = -1e30f;
            }
            sv[j][r] = s;
        }
    }
    // softmax per row (reduce over 16-lane column group; half groups independent)
    #pragma unroll
    for (int r = 0; r < 8; ++r) {
        float mx = fmaxf(fmaxf(sv[0][r], sv[1][r]), fmaxf(sv[2][r], sv[3][r]));
        #pragma unroll
        for (int m = 1; m < 16; m <<= 1) mx = fmaxf(mx, __shfl_xor(mx, m, 32));
        float sum = 0.0f;
        #pragma unroll
        for (int j = 0; j < 4; ++j) { float e = __expf(sv[j][r] - mx); sv[j][r] = e; sum += e; }
        #pragma unroll
        for (int m = 1; m < 16; m <<= 1) sum += __shfl_xor(sum, m, 32);
        float inv = 1.0f / (sum + 1e-20f);
        #pragma unroll
        for (int j = 0; j < 4; ++j) sv[j][r] *= inv;
    }
    // stage P (16x64 f16) in LDS to re-shape D-layout -> A-fragment layout
    #pragma unroll
    for (int j = 0; j < 4; ++j)
        #pragma unroll
        for (int r = 0; r < 8; ++r)
            plds[wave][(r + 8 * hh) * 64 + j * 16 + l16] = (_Float16)sv[j][r];
    __syncthreads();

    v16h ap[2];
    #pragma unroll
    for (int s = 0; s < 2; ++s)
        #pragma unroll
        for (int e = 0; e < 16; ++e)
            ap[s][e] = plds[wave][l16 * 64 + 32 * s + a_kofs(e, hh)];

    // O = P V  (2 column tiles of 16, K=64 in 2 steps) — V fragments from vt, contiguous
    #pragma unroll
    for (int n = 0; n < 2; ++n) {
        v8f acc = {};
        #pragma unroll
        for (int s = 0; s < 2; ++s) {
            v16h bv;
            const _Float16* vr = vt + (head * 32 + n * 16 + l16) * 64 + 32 * s;
            #pragma unroll
            for (int e = 0; e < 16; ++e) bv[e] = vr[b_kofs(e, hh)];
            acc = WMMA_F16(ap[s], bv, acc);
        }
        #pragma unroll
        for (int r = 0; r < 8; ++r) {
            int tok = m0 + r + 8 * hh;
            if (tok < NTK)
                attn_out[((long)w * NTK + tok) * CC + head * 32 + n * 16 + l16] = (_Float16)acc[r];
        }
    }
}

extern "C" void kernel_launch(void* const* d_in, const int* in_sizes, int n_in,
                              void* d_out, int out_size, void* d_ws, size_t ws_size,
                              hipStream_t stream) {
    const float* x       = (const float*)d_in[0];
    const float* norm1_g = (const float*)d_in[1];
    const float* norm1_b = (const float*)d_in[2];
    const float* qkv_w   = (const float*)d_in[3];
    const float* qkv_b   = (const float*)d_in[4];
    const float* rel_b   = (const float*)d_in[5];
    const float* proj_w  = (const float*)d_in[6];
    const float* proj_b  = (const float*)d_in[7];
    const float* norm2_g = (const float*)d_in[8];
    const float* norm2_b = (const float*)d_in[9];
    const float* fc1_w   = (const float*)d_in[10];
    const float* fc1_b   = (const float*)d_in[11];
    const float* fc2_w   = (const float*)d_in[12];
    const float* fc2_b   = (const float*)d_in[13];
    float* out = (float*)d_out;

    char* ws = (char*)d_ws;
    // region plan (bytes):
    // R0: f16 M*96   (xw -> attn_out -> xn2)          @ 0            38,535,168
    // R1: f16 M*384  (qkv uses first 288 cols -> h)   @ 38,535,168   154,140,672
    // R2: f32 M*96   (x1)                             @ 192,675,840  77,070,336
    // weights f16                                     @ 269,746,176
    _Float16* R0 = (_Float16*)(ws);
    _Float16* R1 = (_Float16*)(ws + 38535168UL);
    float*    R2 = (float*)   (ws + 192675840UL);
    _Float16* w16    = (_Float16*)(ws + 269746176UL);
    _Float16* qkvw16  = w16;                // 288*96
    _Float16* projw16 = qkvw16 + 27648;     // 96*96
    _Float16* fc1w16  = projw16 + 9216;     // 384*96
    _Float16* fc2w16  = fc1w16 + 36864;     // 96*384

    // 1) weight conversions (f32 -> f16)
    cvt_f16_kernel<<<(27648 + 255) / 256, 256, 0, stream>>>(qkv_w, qkvw16, 27648);
    cvt_f16_kernel<<<(9216  + 255) / 256, 256, 0, stream>>>(proj_w, projw16, 9216);
    cvt_f16_kernel<<<(36864 + 255) / 256, 256, 0, stream>>>(fc1_w, fc1w16, 36864);
    cvt_f16_kernel<<<(36864 + 255) / 256, 256, 0, stream>>>(fc2_w, fc2w16, 36864);

    // 2) LN1 + shift + window partition
    ln1_shift_window_kernel<<<MTOT / 8, 256, 0, stream>>>(x, norm1_g, norm1_b, R0);

    const int GB = MTOT / 16 / 8;   // 1568 blocks, 8 waves each

    // 3) QKV GEMM (weights 55,296 B staged once via TDM)
    gemm_kernel<3, QKVN, QKVN, 0><<<GB, 256, QKVN * 96 * 2, stream>>>(R0, qkvw16, qkv_b, nullptr, R1);

    // 4) attention: one block per window (3 heads x 4 strips = 12 waves)
    attn_kernel<<<NWIN, 384, 0, stream>>>(R1, rel_b, R0);

    // 5) proj GEMM + window reverse + unshift + residual -> x1 f32 (weights 18,432 B)
    gemm_kernel<3, 96, 96, 1><<<GB, 256, 96 * 96 * 2, stream>>>(R0, projw16, proj_b, x, R2);

    // 6) LN2 -> xn2 f16 (reuse R0)
    ln2_kernel<<<MTOT / 8, 256, 0, stream>>>(R2, norm2_g, norm2_b, R0);

    // 7) FC1 + GELU -> h f16 (reuse R1); weights staged in 2 phases of 36,864 B
    gemm_kernel<3, HID, HID / 2, 2><<<GB, 256, (HID / 2) * 96 * 2, stream>>>(R0, fc1w16, fc1_b, nullptr, R1);

    // 8) FC2 + residual -> out f32; weights staged in 2 phases of 36,864 B
    gemm_kernel<12, 96, 48, 3><<<GB, 256, 48 * HID * 2, stream>>>(R1, fc2w16, fc2_b, R2, out);

    (void)in_sizes; (void)n_in; (void)out_size; (void)ws_size;
}
